// DeltaNetAttention_73933567033935
// MI455X (gfx1250) — matrix operand order repaired and checked by reading, not compiled, not run on hardware
//
#include <hip/hip_runtime.h>

typedef __attribute__((ext_vector_type(16))) __bf16 v16bf;
typedef __attribute__((ext_vector_type(8)))  __bf16 bf16x8;
typedef __attribute__((ext_vector_type(8)))  float  v8f;
typedef __attribute__((ext_vector_type(4)))  int    int4v;

#define LDA_H 40   // LDS tile row stride in halfs (80 B): conflict-avoiding pad
#define LDC_F 68   // C tile LDS row stride (floats)

typedef __attribute__((address_space(1))) int4v glb_int4;   // "__device__" AS
typedef __attribute__((address_space(3))) int4v lds_int4;   // "__shared__" AS

__device__ __forceinline__ __bf16 f2bf(float f) {
  union { float f; unsigned u; } v; v.f = f;
  unsigned r = v.u + 0x7FFFu + ((v.u >> 16) & 1u);   // round-to-nearest-even
  union { unsigned short s; __bf16 b; } o; o.s = (unsigned short)(r >> 16);
  return o.b;
}

__device__ __forceinline__ v16bf cat8(bf16x8 lo, bf16x8 hi) {
  return __builtin_shufflevector(lo, hi, 0,1,2,3,4,5,6,7,8,9,10,11,12,13,14,15);
}

// 16-byte global -> LDS copy on the CDNA5 async path (ASYNCcnt) when available.
__device__ __forceinline__ void async_copy16(void* lds_dst, const void* gsrc) {
#if __has_builtin(__builtin_amdgcn_global_load_async_to_lds_b128)
  __builtin_amdgcn_global_load_async_to_lds_b128(
      (glb_int4*)gsrc, (lds_int4*)lds_dst, 0, 0);
#else
  *(float4*)lds_dst = *(const float4*)gsrc;
#endif
}
__device__ __forceinline__ void async_wait() {
#if __has_builtin(__builtin_amdgcn_global_load_async_to_lds_b128)
#if __has_builtin(__builtin_amdgcn_s_wait_asynccnt)
  __builtin_amdgcn_s_wait_asynccnt(0);
#else
  asm volatile("s_wait_asynccnt 0x0" ::: "memory");
#endif
#endif
}

// stage 64 rows x 32 halfs (row stride 1024 halfs in global) into LDS tile
__device__ __forceinline__ void stage_rows(__bf16* lds, const __bf16* g, int tid) {
  #pragma unroll
  for (int u = 0; u < 2; ++u) {
    const int c = tid + u * 128;          // 256 16B chunks, 128 threads
    const int r = c >> 2, c4 = c & 3;
    async_copy16(&lds[r * LDA_H + c4 * 8], g + (size_t)r * 1024 + c4 * 8);
  }
}

// ---------------------------------------------------------------------------
// fp32 -> bf16 streaming convert (8 elements / thread)
// ---------------------------------------------------------------------------
__global__ __launch_bounds__(256)
void deltanet_cvt_bf16(const float* __restrict__ src, __bf16* __restrict__ dst) {
  const size_t i = ((size_t)blockIdx.x * 256 + threadIdx.x) * 8;
  const float4 a = *(const float4*)(src + i);
  const float4 b = *(const float4*)(src + i + 4);
  bf16x8 o;
  o[0] = f2bf(a.x); o[1] = f2bf(a.y); o[2] = f2bf(a.z); o[3] = f2bf(a.w);
  o[4] = f2bf(b.x); o[5] = f2bf(b.y); o[6] = f2bf(b.z); o[7] = f2bf(b.w);
  *(bf16x8*)(dst + i) = o;
}

// ---------------------------------------------------------------------------
// Tiled GEMM  C[64x64] = A[64xK] * W[NxK]^T  (K=1024), bf16 in, f32 accum.
// Double-buffered async global->LDS staging.
// mode 0: per-head LayerNorm epilogue, bf16 out     (Q path)
// mode 1: LayerNorm + L2-normalize, f32 out         (K path)
// mode 2: plain f32 out                             (V path)
// mode 3: +bias, f32 out                            (Wo path)
// ---------------------------------------------------------------------------
__global__ __launch_bounds__(128)
void deltanet_gemm_bf16(const __bf16* __restrict__ A,
                        const __bf16* __restrict__ W,
                        int mode,
                        const float* __restrict__ gamma,
                        const float* __restrict__ betaLN,
                        const float* __restrict__ bias,
                        __bf16* __restrict__ out_bf16,
                        float*  __restrict__ out_f32)
{
  __shared__ __attribute__((aligned(16))) __bf16 sAbuf[2][64 * LDA_H];
  __shared__ __attribute__((aligned(16))) __bf16 sBbuf[2][64 * LDA_H];
  __shared__ __attribute__((aligned(16))) float  sC[64 * LDC_F];

  const int tid  = threadIdx.x;
  const int wave = tid >> 5;
  const int lane = tid & 31;
  const int half = lane >> 4;
  const int l16  = lane & 15;

  const int m0 = blockIdx.x * 64;
  const int n0 = blockIdx.y * 64;

  const __bf16* Ab = A + (size_t)m0 * 1024;
  const __bf16* Bb = W + (size_t)n0 * 1024;

  v8f acc[4] = {};

  stage_rows(sAbuf[0], Ab, tid);
  stage_rows(sBbuf[0], Bb, tid);

  int buf = 0;
  const int arow = wave * 16 + l16;
  const int ak   = half * 8;

  for (int kb = 0; kb < 1024; kb += 32) {
    async_wait();
    __syncthreads();
    if (kb + 32 < 1024) {                 // prefetch next chunk into other buffer
      stage_rows(sAbuf[buf ^ 1], Ab + kb + 32, tid);
      stage_rows(sBbuf[buf ^ 1], Bb + kb + 32, tid);
    }
    const __bf16* pA = sAbuf[buf];
    const __bf16* pB = sBbuf[buf];

    // A fragment: lanes<16 hold K {0..7,16..23}, lanes>=16 {8..15,24..31}
    const bf16x8 a_lo = *(const bf16x8*)(&pA[arow * LDA_H + ak]);
    const bf16x8 a_hi = *(const bf16x8*)(&pA[arow * LDA_H + ak + 16]);
    const v16bf af = cat8(a_lo, a_hi);

    #pragma unroll
    for (int nb = 0; nb < 4; ++nb) {
      const int bn = nb * 16 + l16;       // B: same K order as A fragment
      const int bk = half * 8;
      const bf16x8 b_lo = *(const bf16x8*)(&pB[bn * LDA_H + bk]);
      const bf16x8 b_hi = *(const bf16x8*)(&pB[bn * LDA_H + bk + 16]);
      acc[nb] = __builtin_amdgcn_wmma_f32_16x16x32_bf16(
          false, af, false, cat8(b_lo, b_hi), (short)0, acc[nb], false, false);
    }
    buf ^= 1;
  }

  __syncthreads();
  #pragma unroll
  for (int nb = 0; nb < 4; ++nb)
    #pragma unroll
    for (int r = 0; r < 8; ++r)
      sC[(wave * 16 + r + 8 * half) * LDC_F + nb * 16 + l16] = acc[nb][r];
  __syncthreads();

  if (mode == 0 || mode == 1) {
    if (tid < 64) {                       // per-head LN over 64-wide tile row
      float* row = &sC[tid * LDC_F];
      float mu = 0.f;
      #pragma unroll
      for (int j = 0; j < 64; ++j) mu += row[j];
      mu *= (1.f / 64.f);
      float var = 0.f;
      #pragma unroll
      for (int j = 0; j < 64; ++j) { const float d = row[j] - mu; var += d * d; }
      var *= (1.f / 64.f);
      const float inv = rsqrtf(var + 1e-5f);
      if (mode == 0) {
        #pragma unroll
        for (int j = 0; j < 64; ++j)
          row[j] = (row[j] - mu) * inv * gamma[j] + betaLN[j];
      } else {
        float n2 = 0.f;
        #pragma unroll
        for (int j = 0; j < 64; ++j) {
          const float val = (row[j] - mu) * inv * gamma[j] + betaLN[j];
          row[j] = val; n2 += val * val;
        }
        const float s = 1.f / fmaxf(sqrtf(n2), 1e-12f);
        #pragma unroll
        for (int j = 0; j < 64; ++j) row[j] *= s;
      }
    }
  } else if (mode == 3) {
    for (int i = tid; i < 4096; i += 128) {
      const int r = i >> 6, j = i & 63;
      sC[r * LDC_F + j] += bias[n0 + j];
    }
  }
  __syncthreads();

  if (mode == 0) {
    for (int i = tid; i < 4096; i += 128) {
      const int r = i >> 6, j = i & 63;
      out_bf16[(size_t)(m0 + r) * 1024 + n0 + j] = f2bf(sC[r * LDC_F + j]);
    }
  } else {
    for (int i = tid; i < 4096; i += 128) {
      const int r = i >> 6, j = i & 63;
      out_f32[(size_t)(m0 + r) * 1024 + n0 + j] = sC[r * LDC_F + j];
    }
  }
}

// ---------------------------------------------------------------------------
// beta/alpha sigmoid gates: [B*Tkv, H] each
// ---------------------------------------------------------------------------
__global__ __launch_bounds__(256)
void deltanet_gate(const float* __restrict__ keyx,
                   const float* __restrict__ Wb, const float* __restrict__ bb,
                   const float* __restrict__ Wa, const float* __restrict__ ba,
                   float* __restrict__ beta, float* __restrict__ alpha)
{
  const int total = 4 * 4096 * 16;
  int idx = blockIdx.x * 256 + threadIdx.x;
  if (idx >= 2 * total) return;
  const int sel = idx / total;
  const int r   = idx - sel * total;
  const int token = r >> 4;
  const int h     = r & 15;
  const float* W = (sel ? Wa : Wb) + (size_t)h * 1024;
  const float* x = keyx + (size_t)token * 1024;
  float s = 0.f;
  for (int c = 0; c < 1024; c += 4) {
    const float4 xf = *(const float4*)(x + c);
    const float4 wf = *(const float4*)(W + c);
    s += xf.x * wf.x + xf.y * wf.y + xf.z * wf.z + xf.w * wf.w;
  }
  s += (sel ? ba[h] : bb[h]);
  const float g = 1.f / (1.f + __expf(-s));
  (sel ? alpha : beta)[r] = g;
}

// ---------------------------------------------------------------------------
// Sequential delta-rule scan: one workgroup per (b,h); M rows live in VGPRs.
// thread (row = tid/4, quad = tid%4) owns M[row][16*quad .. +15].
// ---------------------------------------------------------------------------
__global__ __launch_bounds__(256)
void deltanet_scan(const float* __restrict__ kn, const float* __restrict__ vv,
                   const float* __restrict__ beta, const float* __restrict__ alpha,
                   float* __restrict__ Mst)
{
  __shared__ float ks[64], vs[64], sc[2];
  const int bh = blockIdx.x;
  const int b = bh >> 4, h = bh & 15;
  const int tid = threadIdx.x;
  const int row = tid >> 2;
  const int c0  = (tid & 3) * 16;

  float m[16];
  #pragma unroll
  for (int e = 0; e < 16; ++e) m[e] = 0.f;

  const size_t tokbase = (size_t)b * 4096;
  for (int t = 0; t < 4096; ++t) {
    const size_t off = (tokbase + t) * 1024 + h * 64;
    __syncthreads();
    if (tid < 64)            ks[tid]      = kn[off + tid];
    else if (tid < 128)      vs[tid - 64] = vv[off + tid - 64];
    else if (tid == 128) {
      const size_t so = (tokbase + t) * 16 + h;
      sc[0] = beta[so]; sc[1] = alpha[so];
    }
    __syncthreads();

    float p = 0.f;                       // partial of (M k)[row] over quad slice
    #pragma unroll
    for (int e = 0; e < 16; ++e) p += m[e] * ks[c0 + e];
    p += __shfl_xor(p, 1, 32);
    p += __shfl_xor(p, 2, 32);           // full dot in all 4 quad lanes
    const float coef = sc[0] * vs[row] - sc[1] * p;   // beta*v[i] - alpha*(Mk)[i]
    #pragma unroll
    for (int e = 0; e < 16; ++e) m[e] += coef * ks[c0 + e];
  }

  float* out = Mst + (size_t)bh * 4096 + row * 64 + c0;
  #pragma unroll
  for (int e4 = 0; e4 < 4; ++e4)
    *(float4*)(out + e4 * 4) = make_float4(m[e4*4], m[e4*4+1], m[e4*4+2], m[e4*4+3]);
}

// ---------------------------------------------------------------------------
// attn[t,i] = sum_j M[i][j] * q[t][j] per (b,h): K=64 bf16 WMMA; bf16 output.
// ---------------------------------------------------------------------------
__global__ __launch_bounds__(128)
void deltanet_qM(const __bf16* __restrict__ qn,
                 const float* __restrict__ Mst,
                 __bf16* __restrict__ attnb)
{
  __shared__ __attribute__((aligned(16))) __bf16 sA[64 * 72];
  __shared__ __attribute__((aligned(16))) __bf16 sB[64 * 72];
  __shared__ __attribute__((aligned(16))) float  sC[64 * LDC_F];

  const int tid  = threadIdx.x;
  const int wave = tid >> 5;
  const int lane = tid & 31;
  const int half = lane >> 4;
  const int l16  = lane & 15;

  const int bh = blockIdx.y;
  const int b = bh >> 4, h = bh & 15;
  const size_t rowbase = (size_t)b * 2048 + blockIdx.x * 64;
  const size_t qbase   = rowbase * 1024 + h * 64;

  // stage q rows (bf16) via async path: 64 rows x 64 halfs
  #pragma unroll
  for (int u = 0; u < 4; ++u) {
    const int c = tid + u * 128;
    const int r = c >> 3, c8 = c & 7;
    async_copy16(&sA[r * 72 + c8 * 8], qn + qbase + (size_t)r * 1024 + c8 * 8);
  }
  // stage M (fp32 -> bf16): row-major M[i][j] is exactly the LDS B layout
  const float* Mp = Mst + (size_t)bh * 4096;
  for (int i = tid; i < 1024; i += 128) {
    const int r = i >> 4, c4 = i & 15;
    const float4 f = *(const float4*)(Mp + r * 64 + c4 * 4);
    __bf16* d = &sB[r * 72 + c4 * 4];
    d[0] = f2bf(f.x); d[1] = f2bf(f.y); d[2] = f2bf(f.z); d[3] = f2bf(f.w);
  }
  async_wait();
  __syncthreads();

  v8f acc[4] = {};
  const int arow = wave * 16 + l16;
  const int ak   = half * 8;
  #pragma unroll
  for (int kb = 0; kb < 64; kb += 32) {
    const bf16x8 a_lo = *(const bf16x8*)(&sA[arow * 72 + kb + ak]);
    const bf16x8 a_hi = *(const bf16x8*)(&sA[arow * 72 + kb + ak + 16]);
    const v16bf af = cat8(a_lo, a_hi);
    #pragma unroll
    for (int nb = 0; nb < 4; ++nb) {
      const int bn = nb * 16 + l16;
      const bf16x8 b_lo = *(const bf16x8*)(&sB[bn * 72 + kb + half * 8]);
      const bf16x8 b_hi = *(const bf16x8*)(&sB[bn * 72 + kb + half * 8 + 16]);
      acc[nb] = __builtin_amdgcn_wmma_f32_16x16x32_bf16(
          false, af, false, cat8(b_lo, b_hi), (short)0, acc[nb], false, false);
    }
  }

  __syncthreads();
  #pragma unroll
  for (int nb = 0; nb < 4; ++nb)
    #pragma unroll
    for (int r = 0; r < 8; ++r)
      sC[(wave * 16 + r + 8 * half) * LDC_F + nb * 16 + l16] = acc[nb][r];
  __syncthreads();

  for (int i = tid; i < 4096; i += 128) {
    const int r = i >> 6, j = i & 63;
    attnb[(rowbase + r) * 1024 + h * 64 + j] = f2bf(sC[r * LDC_F + j]);
  }
}

// ---------------------------------------------------------------------------
extern "C" void kernel_launch(void* const* d_in, const int* in_sizes, int n_in,
                              void* d_out, int out_size, void* d_ws, size_t ws_size,
                              hipStream_t stream) {
  const float* query  = (const float*)d_in[0];
  const float* keyx   = (const float*)d_in[1];
  const float* value  = (const float*)d_in[2];
  const float* Wq     = (const float*)d_in[3];
  const float* Wk     = (const float*)d_in[4];
  const float* Wv     = (const float*)d_in[5];
  const float* Wo     = (const float*)d_in[6];
  const float* bo     = (const float*)d_in[7];
  const float* Wbeta  = (const float*)d_in[8];
  const float* bbeta  = (const float*)d_in[9];
  const float* Walpha = (const float*)d_in[10];
  const float* balpha = (const float*)d_in[11];
  const float* gq     = (const float*)d_in[12];
  const float* bq     = (const float*)d_in[13];
  const float* gk     = (const float*)d_in[14];
  const float* bk     = (const float*)d_in[15];

  // workspace carve (every region fully written before read; deterministic)
  char* w = (char*)d_ws;
  __bf16* qbf  = (__bf16*)w; w += (size_t)4 * 2048 * 1024 * 2;   // 16 MB
  __bf16* kbf  = (__bf16*)w; w += (size_t)4 * 4096 * 1024 * 2;   // 32 MB
  __bf16* vbf  = (__bf16*)w; w += (size_t)4 * 4096 * 1024 * 2;   // 32 MB
  __bf16* Wqb  = (__bf16*)w; w += (size_t)1024 * 1024 * 2;       //  2 MB
  __bf16* Wkb  = (__bf16*)w; w += (size_t)1024 * 1024 * 2;
  __bf16* Wvb  = (__bf16*)w; w += (size_t)1024 * 1024 * 2;
  __bf16* Wob  = (__bf16*)w; w += (size_t)1024 * 1024 * 2;
  __bf16* qn   = (__bf16*)w; w += (size_t)4 * 2048 * 1024 * 2;   // 16 MB
  float*  kn   = (float*)w;  w += (size_t)4 * 4096 * 1024 * 4;   // 64 MB
  float*  vv   = (float*)w;  w += (size_t)4 * 4096 * 1024 * 4;   // 64 MB
  float*  bet  = (float*)w;  w += (size_t)4 * 4096 * 16 * 4;     //  1 MB
  float*  alp  = (float*)w;  w += (size_t)4 * 4096 * 16 * 4;     //  1 MB
  float*  Mst  = (float*)w;  w += (size_t)64 * 64 * 64 * 4;      //  1 MB
  __bf16* attb = (__bf16*)w; w += (size_t)4 * 2048 * 1024 * 2;   // 16 MB

  const dim3 blk128(128), blk256(256);

  // precision pre-pass: fp32 -> bf16 (streams once; halves GEMM traffic)
  deltanet_cvt_bf16<<<dim3(4096), blk256, 0, stream>>>(query, qbf);
  deltanet_cvt_bf16<<<dim3(8192), blk256, 0, stream>>>(keyx,  kbf);
  deltanet_cvt_bf16<<<dim3(8192), blk256, 0, stream>>>(value, vbf);
  deltanet_cvt_bf16<<<dim3(512),  blk256, 0, stream>>>(Wq, Wqb);
  deltanet_cvt_bf16<<<dim3(512),  blk256, 0, stream>>>(Wk, Wkb);
  deltanet_cvt_bf16<<<dim3(512),  blk256, 0, stream>>>(Wv, Wvb);
  deltanet_cvt_bf16<<<dim3(512),  blk256, 0, stream>>>(Wo, Wob);

  // Q = LN(query @ Wq^T)          -> bf16 [8192 x 1024]
  deltanet_gemm_bf16<<<dim3(128, 16), blk128, 0, stream>>>(
      qbf, Wqb, 0, gq, bq, nullptr, qn, nullptr);
  // K = normalize(LN(key @ Wk^T)) -> f32 [16384 x 1024]
  deltanet_gemm_bf16<<<dim3(256, 16), blk128, 0, stream>>>(
      kbf, Wkb, 1, gk, bk, nullptr, nullptr, kn);
  // V = value @ Wv^T              -> f32 [16384 x 1024]
  deltanet_gemm_bf16<<<dim3(256, 16), blk128, 0, stream>>>(
      vbf, Wvb, 2, nullptr, nullptr, nullptr, nullptr, vv);
  // gates
  deltanet_gate<<<dim3((2 * 4 * 4096 * 16 + 255) / 256), blk256, 0, stream>>>(
      keyx, Wbeta, bbeta, Walpha, balpha, bet, alp);
  // sequential delta-rule scan -> M [64 x 64 x 64]
  deltanet_scan<<<dim3(64), blk256, 0, stream>>>(kn, vv, bet, alp, Mst);
  // attn = q @ M^T per (b,h) -> bf16 [8192 x 1024]
  deltanet_qM<<<dim3(32, 64), blk128, 0, stream>>>(qn, Mst, attb);
  // out = attn @ Wo^T + bo -> d_out f32
  deltanet_gemm_bf16<<<dim3(128, 16), blk128, 0, stream>>>(
      attb, Wob, 3, nullptr, nullptr, bo, nullptr, (float*)d_out);
}
